// DeepseekV3MoEToA2AAdapter_34918084116587
// MI455X (gfx1250) — matrix operand order repaired and checked by reading, not compiled, not run on hardware
//
#include <hip/hip_runtime.h>
#include <hip/hip_bf16.h>

// ---------------- problem constants ----------------
#define N_EXPERTS 16
#define TOPK      4
#define HIDDEN    1024
#define INTER     512
#define NTOK      2048            // B*S = 2*1024

typedef __attribute__((ext_vector_type(16))) __bf16    v16bf;
typedef __attribute__((ext_vector_type(8)))  float     v8f;
typedef __attribute__((ext_vector_type(4)))  unsigned  u32x4;
typedef __attribute__((ext_vector_type(8)))  unsigned  u32x8;

union Frag {           // 32 bytes: one WMMA bf16 A/B fragment
    uint4 q[2];
    v16bf v;
};

__device__ __forceinline__ unsigned short f2bf(float f) {
    unsigned int u = __builtin_bit_cast(unsigned int, f);
    unsigned int r = u + 0x7FFFu + ((u >> 16) & 1u);   // round-to-nearest-even
    return (unsigned short)(r >> 16);
}

// ---------------- TDM: transposing 32(k) x 64(n) bf16 tile -> LDS [n][k] ----------------
// D# per CDNA5 ISA ch.8: tile_dim0=1 (X), tile_dim1=32 (Y = k, stride = row_stride),
// tile_dim2=64 (Z = n, stride = 1). LDS fill order is sequential => n-major [n][k],
// 64-byte rows: exactly the layout the B-fragment builder reads with ds_load_b128.
__device__ __forceinline__ void tdm_load_b_tile_T(unsigned lds_byte_addr,
                                                  const unsigned short* gsrc,
                                                  unsigned row_stride_elems) {
    const unsigned long long ga = (unsigned long long)(const void*)gsrc;

    u32x4 g0;
    g0.x = 1u;                                   // count=1, is_restore=0, gather off
    g0.y = lds_byte_addr;                        // lds_addr
    g0.z = (unsigned)(ga & 0xFFFFFFFFu);         // global_addr[31:0]
    g0.w = (unsigned)((ga >> 32) & 0x01FFFFFFu)  // global_addr[56:32]
         | (2u << 30);                           // type = 2 ("image")

    u32x8 g1;
    g1.s0 = (1u << 16);                          // workgroup_mask=0, data_size=1 (2 bytes)
    g1.s1 = (0xFFFFu << 16);                     // atomic_barrier=0 | tensor_dim0[15:0]
    g1.s2 = 0x0000u | (0xFFFFu << 16);           // tensor_dim0[31:16]=0 | tensor_dim1[15:0]
    g1.s3 = 0x000Fu | (1u << 16);                // tensor_dim1[31:16] | tile_dim0=1
    g1.s4 = 32u | (64u << 16);                   // tile_dim1=32 (k) | tile_dim2=64 (n)
    g1.s5 = row_stride_elems;                    // tensor_dim0_stride[31:0]
    g1.s6 = 0u | (1u << 16);                     // dim0_stride[47:32]=0 | dim1_stride[15:0]=1
    g1.s7 = 0u;                                  // dim1_stride[47:16]=0

    u32x4 g2;
    g2.x = 0xFFFFu;                              // tensor_dim2 (large, no OOB)
    g2.y = 0u;                                   // tensor_dim3
    g2.z = 0u;                                   // tensor_dim2_stride[31:0]
    g2.w = 0u;                                   // dim2_stride[47:32] | tile_dim3=0

    u32x4 g3;
    g3.x = 0u; g3.y = 0u; g3.z = 0u; g3.w = 0u;  // tensor_dim3_stride / dim4 / tile_dim4

    asm volatile("tensor_load_to_lds %0, %1, %2, %3"
                 :
                 : "s"(g0), "s"(g1), "s"(g2), "s"(g3)
                 : "memory");
}

// ---------------- 0: zero the expert counters ----------------
__global__ void moe_zero_counts(int* counts) {
    if (threadIdx.x < N_EXPERTS) counts[threadIdx.x] = 0;
}

// ---------------- 1: fp32 -> bf16 conversion ----------------
__global__ void moe_cvt_bf16(const float* __restrict__ src,
                             unsigned short* __restrict__ dst, int n) {
    int i = blockIdx.x * blockDim.x + threadIdx.x;
    if (i < n) dst[i] = f2bf(src[i]);
}

// ---------------- 2: router (one wave per token) ----------------
__global__ void moe_router(const float* __restrict__ x,
                           const float* __restrict__ gw,
                           int*   __restrict__ topk_idx,
                           float* __restrict__ topk_wt,
                           int*   __restrict__ counts,
                           int*   __restrict__ tok_list) {
    const int wave = (blockIdx.x * blockDim.x + threadIdx.x) >> 5;
    const int lane = threadIdx.x & 31;
    if (wave >= NTOK) return;
    const float* xr = x + (size_t)wave * HIDDEN;

    float acc[N_EXPERTS];
#pragma unroll
    for (int e = 0; e < N_EXPERTS; ++e) acc[e] = 0.f;

    for (int it = 0; it < HIDDEN / 32; ++it) {
        const int h = lane + it * 32;
        const float xv = xr[h];
#pragma unroll
        for (int e = 0; e < N_EXPERTS; ++e)
            acc[e] += xv * gw[e * HIDDEN + h];
    }
#pragma unroll
    for (int off = 16; off > 0; off >>= 1) {
#pragma unroll
        for (int e = 0; e < N_EXPERTS; ++e)
            acc[e] += __shfl_xor(acc[e], off, 32);
    }

    if (lane == 0) {
        float mx = acc[0];
#pragma unroll
        for (int e = 1; e < N_EXPERTS; ++e) mx = fmaxf(mx, acc[e]);
        float p[N_EXPERTS];
#pragma unroll
        for (int e = 0; e < N_EXPERTS; ++e) p[e] = __expf(acc[e] - mx);

        int   bi[TOPK];
        float bw[TOPK];
        float wsum = 0.f;
#pragma unroll
        for (int k = 0; k < TOPK; ++k) {
            int   best = 0;
            float bv   = -1.f;
#pragma unroll
            for (int e = 0; e < N_EXPERTS; ++e)
                if (p[e] > bv) { bv = p[e]; best = e; }   // stable: first max wins
            bi[k] = best; bw[k] = bv; wsum += bv;
            p[best] = -2.f;
        }
        const float inv = 1.f / wsum;
#pragma unroll
        for (int k = 0; k < TOPK; ++k) {
            topk_idx[wave * TOPK + k] = bi[k];
            topk_wt [wave * TOPK + k] = bw[k] * inv;
            int pos = atomicAdd(&counts[bi[k]], 1);
            tok_list[bi[k] * NTOK + pos] = wave * TOPK + k;   // slot id r = t*4+k
        }
    }
}

// ---------------- 3: stage-1 grouped GEMM: act = silu(x*w1) * (x*w3) ----------------
// block = 256 threads (8 waves, 4x2 wave grid), block tile M=64 x N=64, K-step 32.
// Each wave: 16(M) x 32(N) strip = 2 accumulators per matrix = 4 WMMAs per K-step.
// B tiles are DMAed (and transposed) into LDS by the Tensor Data Mover.
__global__ void moe_gemm1(const unsigned short* __restrict__ xbf,
                          const unsigned short* __restrict__ w1bf,
                          const unsigned short* __restrict__ w3bf,
                          unsigned short* __restrict__ actbf,
                          const int* __restrict__ counts,
                          const int* __restrict__ tok_list) {
    const int e     = blockIdx.z;
    const int mtile = blockIdx.y;
    const int cnt   = counts[e];
    if (mtile * 64 >= cnt) return;
    const int n0 = blockIdx.x * 64;

    __shared__ __align__(16) unsigned short As[64][32];
    __shared__ __align__(16) unsigned short B1[64][32];   // [n][k] (written by TDM)
    __shared__ __align__(16) unsigned short B3[64][32];   // [n][k] (written by TDM)
    __shared__ int rows[64];

    const int tid = threadIdx.x;
    if (tid < 64) {
        const int m = mtile * 64 + tid;
        rows[tid] = (m < cnt) ? tok_list[e * NTOK + m] : -1;
    }
    __syncthreads();

    const int wave = tid >> 5, lane = tid & 31;
    const int wm = wave >> 1, wn = wave & 1;      // 4 x 2 wave grid

    v8f accg[2], accu[2];
#pragma unroll
    for (int j = 0; j < 2; ++j) {
        accg[j] = (v8f){0.f, 0.f, 0.f, 0.f, 0.f, 0.f, 0.f, 0.f};
        accu[j] = (v8f){0.f, 0.f, 0.f, 0.f, 0.f, 0.f, 0.f, 0.f};
    }

    const size_t   wbase   = (size_t)e * HIDDEN * INTER;
    const unsigned lds_b1  = (unsigned)(uintptr_t)&B1[0][0];
    const unsigned lds_b3  = (unsigned)(uintptr_t)&B3[0][0];

    for (int k0 = 0; k0 < HIDDEN; k0 += 32) {
        // ---- kick off the TDM transposing loads of both B tiles (wave 0 only) ----
        if (wave == 0) {
            tdm_load_b_tile_T(lds_b1, w1bf + wbase + (size_t)k0 * INTER + n0, INTER);
            tdm_load_b_tile_T(lds_b3, w3bf + wbase + (size_t)k0 * INTER + n0, INTER);
        }
        // ---- stage A tile (gathered token rows, zero-padded) meanwhile ----
        {
            const int m = tid >> 2, seg = (tid & 3) * 8;
            const int r = rows[m];
            uint4 val = make_uint4(0u, 0u, 0u, 0u);
            if (r >= 0) {
                const int t = r >> 2;
                val = *(const uint4*)(xbf + (size_t)t * HIDDEN + k0 + seg);
            }
            *(uint4*)&As[m][seg] = val;
        }
        if (wave == 0) __builtin_amdgcn_s_wait_tensorcnt(0);
        __syncthreads();

        // ---- build fragments, k = 16*(lane/16) + s in A and B (contiguous order) ----
        Frag a, b1[2], b3[2];
        {
            const int row   = wm * 16 + (lane & 15);
            const int khalf = lane >> 4;
            const uint4* rp = (const uint4*)&As[row][0];
            a.q[0] = rp[2 * khalf];
            a.q[1] = rp[2 * khalf + 1];
        }
        const int kg = lane >> 4;
#pragma unroll
        for (int j = 0; j < 2; ++j) {
            const int col = wn * 32 + j * 16 + (lane & 15);
            const uint4* r1 = (const uint4*)&B1[col][0];
            const uint4* r3 = (const uint4*)&B3[col][0];
            b1[j].q[0] = r1[kg * 2]; b1[j].q[1] = r1[kg * 2 + 1];
            b3[j].q[0] = r3[kg * 2]; b3[j].q[1] = r3[kg * 2 + 1];
        }

#pragma unroll
        for (int j = 0; j < 2; ++j) {
            accg[j] = __builtin_amdgcn_wmma_f32_16x16x32_bf16(false, a.v, false, b1[j].v,
                                                              (short)0, accg[j], false, false);
            accu[j] = __builtin_amdgcn_wmma_f32_16x16x32_bf16(false, a.v, false, b3[j].v,
                                                              (short)0, accu[j], false, false);
        }
        __syncthreads();
    }

    // ---- epilogue: silu(g)*u -> bf16 act rows keyed by slot id ----
    const int nloc  = lane & 15;
    const int mhalf = lane >> 4;
#pragma unroll
    for (int j = 0; j < 2; ++j) {
#pragma unroll
        for (int r8 = 0; r8 < 8; ++r8) {
            const int mloc = wm * 16 + r8 + 8 * mhalf;
            const int rr   = rows[mloc];
            if (rr >= 0) {
                const float g = accg[j][r8], u = accu[j][r8];
                const float s = g / (1.f + __expf(-g));
                actbf[(size_t)rr * INTER + n0 + wn * 32 + j * 16 + nloc] = f2bf(s * u);
            }
        }
    }
}

// ---------------- 4: stage-2 grouped GEMM: eout = act * w2 ----------------
__global__ void moe_gemm2(const unsigned short* __restrict__ actbf,
                          const unsigned short* __restrict__ w2bf,
                          float* __restrict__ eout,
                          const int* __restrict__ counts,
                          const int* __restrict__ tok_list) {
    const int e     = blockIdx.z;
    const int mtile = blockIdx.y;
    const int cnt   = counts[e];
    if (mtile * 64 >= cnt) return;
    const int n0 = blockIdx.x * 64;

    __shared__ __align__(16) unsigned short As[64][32];
    __shared__ __align__(16) unsigned short Bs[64][32];   // [n][k] (written by TDM)
    __shared__ int rows[64];

    const int tid = threadIdx.x;
    if (tid < 64) {
        const int m = mtile * 64 + tid;
        rows[tid] = (m < cnt) ? tok_list[e * NTOK + m] : -1;
    }
    __syncthreads();

    const int wave = tid >> 5, lane = tid & 31;
    const int wm = wave >> 1, wn = wave & 1;

    v8f acc[2];
    acc[0] = (v8f){0.f, 0.f, 0.f, 0.f, 0.f, 0.f, 0.f, 0.f};
    acc[1] = (v8f){0.f, 0.f, 0.f, 0.f, 0.f, 0.f, 0.f, 0.f};

    const size_t   wbase  = (size_t)e * INTER * HIDDEN;
    const unsigned lds_bs = (unsigned)(uintptr_t)&Bs[0][0];

    for (int k0 = 0; k0 < INTER; k0 += 32) {
        if (wave == 0)
            tdm_load_b_tile_T(lds_bs, w2bf + wbase + (size_t)k0 * HIDDEN + n0, HIDDEN);
        {
            const int m = tid >> 2, seg = (tid & 3) * 8;
            const int r = rows[m];
            uint4 val = make_uint4(0u, 0u, 0u, 0u);
            if (r >= 0)
                val = *(const uint4*)(actbf + (size_t)r * INTER + k0 + seg);
            *(uint4*)&As[m][seg] = val;
        }
        if (wave == 0) __builtin_amdgcn_s_wait_tensorcnt(0);
        __syncthreads();

        Frag a, b[2];
        {
            const int row   = wm * 16 + (lane & 15);
            const int khalf = lane >> 4;
            const uint4* rp = (const uint4*)&As[row][0];
            a.q[0] = rp[2 * khalf];
            a.q[1] = rp[2 * khalf + 1];
        }
        const int kg = lane >> 4;
#pragma unroll
        for (int j = 0; j < 2; ++j) {
            const int col = wn * 32 + j * 16 + (lane & 15);
            const uint4* rp = (const uint4*)&Bs[col][0];
            b[j].q[0] = rp[kg * 2]; b[j].q[1] = rp[kg * 2 + 1];
        }

#pragma unroll
        for (int j = 0; j < 2; ++j)
            acc[j] = __builtin_amdgcn_wmma_f32_16x16x32_bf16(false, a.v, false, b[j].v,
                                                             (short)0, acc[j], false, false);
        __syncthreads();
    }

    const int nloc  = lane & 15;
    const int mhalf = lane >> 4;
#pragma unroll
    for (int j = 0; j < 2; ++j) {
#pragma unroll
        for (int r8 = 0; r8 < 8; ++r8) {
            const int mloc = wm * 16 + r8 + 8 * mhalf;
            const int rr   = rows[mloc];
            if (rr >= 0)
                eout[(size_t)rr * HIDDEN + n0 + wn * 32 + j * 16 + nloc] = acc[j][r8];
        }
    }
}

// ---------------- 5: combine: out[t] = sum_k w_k * eout[t*4+k] ----------------
__global__ void moe_combine(const float* __restrict__ eout,
                            const float* __restrict__ topk_wt,
                            float* __restrict__ out) {
    const int i = blockIdx.x * blockDim.x + threadIdx.x;
    if (i >= NTOK * HIDDEN) return;
    const int t = i / HIDDEN;
    const int h = i - t * HIDDEN;
    float s = 0.f;
#pragma unroll
    for (int k = 0; k < TOPK; ++k)
        s += topk_wt[t * TOPK + k] * eout[(size_t)(t * TOPK + k) * HIDDEN + h];
    out[i] = s;
}

// ---------------- host-side launcher ----------------
extern "C" void kernel_launch(void* const* d_in, const int* in_sizes, int n_in,
                              void* d_out, int out_size, void* d_ws, size_t ws_size,
                              hipStream_t stream) {
    const float* x  = (const float*)d_in[0];   // [T, H]
    const float* gw = (const float*)d_in[1];   // [E, H]
    const float* w1 = (const float*)d_in[2];   // [E, H, I]
    const float* w3 = (const float*)d_in[3];   // [E, H, I]
    const float* w2 = (const float*)d_in[4];   // [E, I, H]
    float* out = (float*)d_out;                // [T, H]

    char* ws = (char*)d_ws;
    size_t off = 0;
    auto alloc = [&](size_t bytes) {
        char* p = ws + off;
        off += (bytes + 255) & ~(size_t)255;
        return p;
    };
    int*            counts   = (int*)           alloc(N_EXPERTS * sizeof(int));
    int*            topk_idx = (int*)           alloc((size_t)NTOK * TOPK * sizeof(int));
    float*          topk_wt  = (float*)         alloc((size_t)NTOK * TOPK * sizeof(float));
    int*            tok_list = (int*)           alloc((size_t)N_EXPERTS * NTOK * sizeof(int));
    unsigned short* xbf      = (unsigned short*)alloc((size_t)NTOK * HIDDEN * 2);
    unsigned short* w1bf     = (unsigned short*)alloc((size_t)N_EXPERTS * HIDDEN * INTER * 2);
    unsigned short* w3bf     = (unsigned short*)alloc((size_t)N_EXPERTS * HIDDEN * INTER * 2);
    unsigned short* w2bf     = (unsigned short*)alloc((size_t)N_EXPERTS * INTER * HIDDEN * 2);
    unsigned short* actbf    = (unsigned short*)alloc((size_t)NTOK * TOPK * INTER * 2);
    float*          eout     = (float*)         alloc((size_t)NTOK * TOPK * HIDDEN * sizeof(float));
    (void)ws_size; (void)in_sizes; (void)n_in; (void)out_size; (void)topk_idx;

    moe_zero_counts<<<1, 32, 0, stream>>>(counts);

    const int nx = NTOK * HIDDEN;
    const int nw = N_EXPERTS * HIDDEN * INTER;
    moe_cvt_bf16<<<(nx + 255) / 256, 256, 0, stream>>>(x,  xbf,  nx);
    moe_cvt_bf16<<<(nw + 255) / 256, 256, 0, stream>>>(w1, w1bf, nw);
    moe_cvt_bf16<<<(nw + 255) / 256, 256, 0, stream>>>(w3, w3bf, nw);
    moe_cvt_bf16<<<(nw + 255) / 256, 256, 0, stream>>>(w2, w2bf, nw);

    moe_router<<<NTOK / 8, 256, 0, stream>>>(x, gw, topk_idx, topk_wt, counts, tok_list);

    moe_gemm1<<<dim3(INTER / 64, NTOK / 64, N_EXPERTS), 256, 0, stream>>>(
        xbf, w1bf, w3bf, actbf, counts, tok_list);
    moe_gemm2<<<dim3(HIDDEN / 64, NTOK / 64, N_EXPERTS), 256, 0, stream>>>(
        actbf, w2bf, eout, counts, tok_list);

    moe_combine<<<(NTOK * HIDDEN + 255) / 256, 256, 0, stream>>>(eout, topk_wt, out);
}
